// ContrastiveHGNN_84335977824259
// MI455X (gfx1250) — matrix operand order repaired and checked by reading, not compiled, not run on hardware
//
#include <hip/hip_runtime.h>
#include <hip/hip_bf16.h>

// Problem constants (from the reference)
#define NN   4096
#define EE   4096
#define BB   3
#define DIN  64
#define HID  32
#define OUTD 32
#define NLAY 2
#define EPSV 1e-6f

typedef __attribute__((ext_vector_type(16))) _Float16 v16h;
typedef __attribute__((ext_vector_type(8)))  _Float16 v8h;
typedef __attribute__((ext_vector_type(8)))  float    v8f;

union V16 { v16h v; v8h h[2]; };

// ---------------------------------------------------------------------------
// 1) Convert H (fp32, binary) -> Hh fp16 row-major AND HhT fp16 (transposed).
//    Binary values are exact in fp16; halves H footprint so Hh+HhT (192MB)
//    is L2-resident on MI455X (192MB L2) for all subsequent GEMM passes.
// ---------------------------------------------------------------------------
__global__ __launch_bounds__(256) void k_convert(const float* __restrict__ H,
                                                 _Float16* __restrict__ Hh,
                                                 _Float16* __restrict__ HhT) {
  __shared__ _Float16 tile[32][33];
  const int b  = blockIdx.z;
  const int e0 = blockIdx.x * 32;
  const int n0 = blockIdx.y * 32;
  const float*    Hb   = H   + (size_t)b * NN * EE;
  _Float16*       Hhb  = Hh  + (size_t)b * NN * EE;
  _Float16*       HhTb = HhT + (size_t)b * EE * NN;
  const int tx = threadIdx.x, ty = threadIdx.y;
  #pragma unroll
  for (int r = 0; r < 4; ++r) {
    const int nl = ty + r * 8;
    _Float16 h = (_Float16)Hb[(size_t)(n0 + nl) * EE + e0 + tx];
    Hhb[(size_t)(n0 + nl) * EE + e0 + tx] = h;     // coalesced
    tile[nl][tx] = h;
  }
  __syncthreads();
  #pragma unroll
  for (int r = 0; r < 4; ++r) {
    const int el = ty + r * 8;
    HhTb[(size_t)(e0 + el) * NN + n0 + tx] = tile[tx][el];  // coalesced
  }
}

// dv[b,n] = rsqrt(max(sum_e H, eps))  — reads fp16 Hh (half the bytes)
__global__ __launch_bounds__(256) void k_dv(const _Float16* __restrict__ Hh,
                                            float* __restrict__ dv) {
  const int n = blockIdx.x, b = blockIdx.y;
  const _Float16* row = Hh + ((size_t)b * NN + n) * EE;
  float s = 0.f;
  for (int e = threadIdx.x; e < EE; e += 256) s += (float)row[e];
  __shared__ float red[256];
  red[threadIdx.x] = s; __syncthreads();
  for (int st = 128; st > 0; st >>= 1) {
    if (threadIdx.x < st) red[threadIdx.x] += red[threadIdx.x + st];
    __syncthreads();
  }
  if (threadIdx.x == 0) dv[(size_t)b * NN + n] = rsqrtf(fmaxf(red[0], EPSV));
}

// de[b,e] = 1/max(sum_n H, eps)  — thread-per-column, coalesced across lanes
__global__ __launch_bounds__(256) void k_de(const _Float16* __restrict__ Hh,
                                            float* __restrict__ de) {
  const int e = blockIdx.x * 256 + threadIdx.x;
  const int b = blockIdx.y;
  const _Float16* Hb = Hh + (size_t)b * NN * EE;
  float s = 0.f;
  for (int n = 0; n < NN; ++n) s += (float)Hb[(size_t)n * EE + e];
  de[(size_t)b * EE + e] = 1.f / fmaxf(s, EPSV);
}

// x0 = relu(X @ W_init + b_init)   [N, HID]
__global__ __launch_bounds__(256) void k_init(const float* __restrict__ X,
                                              const float* __restrict__ W,
                                              const float* __restrict__ bias,
                                              float* __restrict__ x) {
  const int gid = blockIdx.x * 256 + threadIdx.x;   // over NN*HID
  const int n = gid >> 5, h = gid & 31;
  float s = bias[h];
  #pragma unroll 8
  for (int k = 0; k < DIN; ++k) s += X[(size_t)n * DIN + k] * W[k * HID + h];
  x[gid] = fmaxf(s, 0.f);
}

// xw = x @ W_node[l] + b_node[l];  yT[b][h][n] = fp16(dv[b,n] * xw[n,h])
// yT is transposed so WMMA B fragments are contiguous per lane.
__global__ __launch_bounds__(256) void k_prep(const float* __restrict__ x,
                                              const float* __restrict__ W,
                                              const float* __restrict__ bias,
                                              const float* __restrict__ dv,
                                              _Float16* __restrict__ yT) {
  const int gid = blockIdx.x * 256 + threadIdx.x;   // over HID*NN, n fastest
  const int n = gid & (NN - 1);
  const int h = gid >> 12;
  float s = bias[h];
  #pragma unroll 8
  for (int k = 0; k < HID; ++k) s += x[n * HID + k] * W[k * HID + h];
  #pragma unroll
  for (int b = 0; b < BB; ++b)
    yT[((size_t)b * HID + h) * NN + n] = (_Float16)(dv[(size_t)b * NN + n] * s);
}

// ---------------------------------------------------------------------------
// Core WMMA GEMM:  D[4096,32] = rowScale ⊙ (A[4096,4096] @ Bt^T), per behavior.
// A = Hh or HhT (fp16, row-major, L2-resident). Bt = [32,4096] fp16 transposed
// small operand. Block = 4 waves; each wave takes K/4=1024 (32 x v_wmma
// f32_16x16x32_f16 per output half), partials reduced through LDS.
// MODE 0: scale=de, store fp16 transposed ([32,4096]) feeding the next hop.
// MODE 1: scale=dv, store fp32 u [4096,32] standard layout.
// ---------------------------------------------------------------------------
template <int MODE>
__global__ __launch_bounds__(128) void k_hgemm(const _Float16* __restrict__ Abig,
                                               const _Float16* __restrict__ BtAll,
                                               const float* __restrict__ scaleAll,
                                               float* __restrict__ outF,
                                               _Float16* __restrict__ outT) {
  constexpr int K = 4096, M = 4096;
  const int b = blockIdx.z;
  const _Float16* A     = Abig    + (size_t)b * M * K;
  const _Float16* Bt    = BtAll   + (size_t)b * HID * K;
  const float*    scale = scaleAll + (size_t)b * M;

  const int m0   = blockIdx.x * 16;
  const int lane = threadIdx.x & 31;
  const int wave = threadIdx.x >> 5;
  const int mrow = lane & 15;                 // A: M row per lane
  const int col  = lane & 15;                 // B/D: N column per lane
  const int hi   = lane >> 4;                 // half-wave select
  const int koffA = hi << 4;                  // A: 16 consecutive K per lane
  const int koffB = hi << 4;                  // B: 16 consecutive K per lane

  const int kbeg = wave * (K / 4);
  const _Float16* pA  = A  + (size_t)(m0 + mrow) * K + kbeg + koffA;
  const _Float16* pB0 = Bt + (size_t)col        * K + kbeg + koffB;
  const _Float16* pB1 = Bt + (size_t)(16 + col) * K + kbeg + koffB;

  v8f acc0 = {}; v8f acc1 = {};
  #pragma unroll 4
  for (int k = 0; k < K / 4; k += 32) {
    V16 a, b0, b1;
    a.h[0]  = *(const v8h*)(pA  + k);
    a.h[1]  = *(const v8h*)(pA  + k + 8);
    b0.h[0] = *(const v8h*)(pB0 + k);
    b0.h[1] = *(const v8h*)(pB0 + k + 8);
    b1.h[0] = *(const v8h*)(pB1 + k);
    b1.h[1] = *(const v8h*)(pB1 + k + 8);
    acc0 = __builtin_amdgcn_wmma_f32_16x16x32_f16(false, a.v, false, b0.v,
                                                  (short)0, acc0, false, false);
    acc1 = __builtin_amdgcn_wmma_f32_16x16x32_f16(false, a.v, false, b1.v,
                                                  (short)0, acc1, false, false);
  }

  __shared__ float red[4][2][8][32];
  #pragma unroll
  for (int i = 0; i < 8; ++i) {
    red[wave][0][i][lane] = acc0[i];
    red[wave][1][i][lane] = acc1[i];
  }
  __syncthreads();

  if (wave == 0) {
    const int mbase = m0 + (hi << 3);   // D layout: VGPR i, lane l -> M=i+8*(l>=16)
    #pragma unroll
    for (int t = 0; t < 2; ++t) {
      float f[8];
      #pragma unroll
      for (int i = 0; i < 8; ++i)
        f[i] = (red[0][t][i][lane] + red[1][t][i][lane]) +
               (red[2][t][i][lane] + red[3][t][i][lane]);
      if (MODE == 0) {
        v8h pk;
        #pragma unroll
        for (int i = 0; i < 8; ++i)
          pk[i] = (_Float16)(f[i] * scale[mbase + i]);
        // transposed store: 8 consecutive M elements per lane -> one b128 store
        *(v8h*)(outT + ((size_t)b * HID + t * 16 + col) * M + mbase) = pk;
      } else {
        #pragma unroll
        for (int i = 0; i < 8; ++i)
          outF[(((size_t)b * M) + mbase + i) * HID + t * 16 + col] =
              f[i] * scale[mbase + i];
      }
    }
  }
}

// x_new = relu( sum_b softmax(bi)[b] * (u[b] @ Theta[b]) )
__global__ __launch_bounds__(256) void k_combine(const float* __restrict__ u,
                                                 const float* __restrict__ Theta,
                                                 const float* __restrict__ bimp,
                                                 float* __restrict__ x) {
  const int gid = blockIdx.x * 256 + threadIdx.x;   // over NN*HID
  const int n = gid >> 5, k = gid & 31;
  const float m = fmaxf(bimp[0], fmaxf(bimp[1], bimp[2]));
  const float w0 = __expf(bimp[0] - m), w1 = __expf(bimp[1] - m),
              w2 = __expf(bimp[2] - m);
  const float inv = 1.f / (w0 + w1 + w2);
  const float wv[3] = {w0 * inv, w1 * inv, w2 * inv};
  float acc = 0.f;
  #pragma unroll
  for (int b = 0; b < BB; ++b) {
    const float* ub = u + ((size_t)b * NN + n) * HID;
    const float* Tb = Theta + (size_t)b * HID * HID;
    float s = 0.f;
    #pragma unroll 8
    for (int h = 0; h < HID; ++h) s += ub[h] * Tb[h * HID + k];
    acc += wv[b] * s;
  }
  x[gid] = fmaxf(acc, 0.f);
}

// t = relu(x @ Wp1 + bp1)
__global__ __launch_bounds__(256) void k_proj1(const float* __restrict__ x,
                                               const float* __restrict__ W,
                                               const float* __restrict__ bias,
                                               float* __restrict__ t) {
  const int gid = blockIdx.x * 256 + threadIdx.x;
  const int n = gid >> 5, h = gid & 31;
  float s = bias[h];
  #pragma unroll 8
  for (int k = 0; k < HID; ++k) s += x[n * HID + k] * W[k * HID + h];
  t[gid] = fmaxf(s, 0.f);
}

// out = t @ Wp2 + bp2
__global__ __launch_bounds__(256) void k_proj2(const float* __restrict__ t,
                                               const float* __restrict__ W,
                                               const float* __restrict__ bias,
                                               float* __restrict__ out) {
  const int gid = blockIdx.x * 256 + threadIdx.x;
  const int n = gid >> 5, o = gid & 31;
  float s = bias[o];
  #pragma unroll 8
  for (int h = 0; h < HID; ++h) s += t[n * HID + h] * W[h * OUTD + o];
  out[gid] = s;
}

extern "C" void kernel_launch(void* const* d_in, const int* in_sizes, int n_in,
                              void* d_out, int out_size, void* d_ws, size_t ws_size,
                              hipStream_t stream) {
  const float* X    = (const float*)d_in[0];
  const float* H    = (const float*)d_in[1];
  const float* Wi   = (const float*)d_in[2];
  const float* bi   = (const float*)d_in[3];
  const float* Wn   = (const float*)d_in[4];   // [L,HID,HID]
  const float* bn   = (const float*)d_in[5];   // [L,HID]
  const float* Th   = (const float*)d_in[6];   // [B,HID,HID]
  const float* bimp = (const float*)d_in[7];   // [B]
  const float* Wp1  = (const float*)d_in[8];
  const float* bp1  = (const float*)d_in[9];
  const float* Wp2  = (const float*)d_in[10];
  const float* bp2  = (const float*)d_in[11];
  float* out = (float*)d_out;

  // Workspace carve (all offsets 256B aligned). Total ~206 MB.
  size_t off = 0;
  auto carve = [&](size_t bytes) -> char* {
    char* p = (char*)d_ws + off;
    off = (off + bytes + 255) & ~(size_t)255;
    return p;
  };
  _Float16* Hh  = (_Float16*)carve((size_t)BB * NN * EE * 2);
  _Float16* HhT = (_Float16*)carve((size_t)BB * EE * NN * 2);
  float*    dv  = (float*)carve((size_t)BB * NN * 4);
  float*    de  = (float*)carve((size_t)BB * EE * 4);
  float*    x   = (float*)carve((size_t)NN * HID * 4);
  _Float16* yT  = (_Float16*)carve((size_t)BB * HID * NN * 2);
  _Float16* zT  = (_Float16*)carve((size_t)BB * HID * EE * 2);
  float*    u   = (float*)carve((size_t)BB * NN * HID * 4);
  float*    tb  = (float*)carve((size_t)NN * HID * 4);
  if (off > ws_size) return;   // workspace too small — nothing safe to do

  // 1) H -> fp16 (row-major + transposed), then degree normalizers from Hh
  k_convert<<<dim3(EE / 32, NN / 32, BB), dim3(32, 8), 0, stream>>>(H, Hh, HhT);
  k_dv<<<dim3(NN, BB), 256, 0, stream>>>(Hh, dv);
  k_de<<<dim3(EE / 256, BB), 256, 0, stream>>>(Hh, de);

  // 2) initial transform
  const int EW = (NN * HID) / 256;   // 512 blocks for elementwise kernels
  k_init<<<EW, 256, 0, stream>>>(X, Wi, bi, x);

  // 3) layers: yT -> (HhT@y)*de -> (Hh@z)*dv -> Theta+softmax+relu
  for (int l = 0; l < NLAY; ++l) {
    k_prep<<<EW, 256, 0, stream>>>(x, Wn + (size_t)l * HID * HID,
                                   bn + (size_t)l * HID, dv, yT);
    k_hgemm<0><<<dim3(EE / 16, 1, BB), 128, 0, stream>>>(HhT, yT, de, nullptr, zT);
    k_hgemm<1><<<dim3(NN / 16, 1, BB), 128, 0, stream>>>(Hh, zT, dv, u, nullptr);
    k_combine<<<EW, 256, 0, stream>>>(u, Th, bimp, x);
  }

  // 4) projection head
  k_proj1<<<EW, 256, 0, stream>>>(x, Wp1, bp1, tb);
  k_proj2<<<EW, 256, 0, stream>>>(tb, Wp2, bp2, out);
}
